// DFFN_22978075034268
// MI455X (gfx1250) — matrix operand order repaired and checked by reading, not compiled, not run on hardware
//
#include <hip/hip_runtime.h>
#include <cstdint>
#include <cstddef>

// ---------------------------------------------------------------------------
// Fused DFFN block for MI455X (gfx1250, wave32, WMMA):
//   project_in (GEMM, bf16 WMMA) -> 2x2 patch spectral filter (Hadamard form)
//   -> depthwise 3x3 conv (paired gate/value) -> exact GELU gating
//   -> project_out (GEMM, bf16 WMMA)
// One workgroup = one 16x16 output tile; h kept entirely in LDS (~284KB).
// ---------------------------------------------------------------------------

typedef __attribute__((ext_vector_type(16))) __bf16 v16bf;
typedef __attribute__((ext_vector_type(8)))  float  v8f;

constexpr int NB   = 4;      // batch
constexpr int CIN  = 64;     // DIM
constexpr int HH   = 256;
constexpr int WW   = 256;
constexpr int HID  = 170;    // HIDDEN
constexpr int HIDP = 176;    // padded to 11 WMMA tiles
constexpr int TILE = 16;     // output tile edge
constexpr int REG  = 20;     // patch-aligned halo region edge (tile +1 halo -> 18 -> patch-round 20)
constexpr int RPX  = REG * REG;   // 400 region pixels
constexpr int NPT  = RPX / 16;    // 25 pixel tiles in GEMM1

// ---- LDS layout (dynamic shared) ----
constexpr int OFF_WBIN  = 0;                               // bf16 [HIDP][CIN]  (B of GEMM1, n-major)
constexpr int OFF_WBOUT = OFF_WBIN  + HIDP * CIN * 2;      // bf16 [CIN][HIDP]  (B of GEMM2, n-major)
constexpr int OFF_WDW   = OFF_WBOUT + CIN * HIDP * 2;      // f32  [340*9]
constexpr int OFF_FFT   = OFF_WDW   + 340 * 9 * 4;         // f32  [170*4]
constexpr int OFF_XT    = ((OFF_FFT + 170 * 4 * 4) + 15) & ~15; // bf16 xt[400][64] / gated[256][176]
constexpr int SZ_XT     = (RPX * CIN * 2) > (256 * HIDP * 2) ? (RPX * CIN * 2) : (256 * HIDP * 2);
constexpr int OFF_H     = ((OFF_XT + SZ_XT) + 15) & ~15;        // bf16 h[400][176] / f32 outb[256][64]
constexpr int SZ_H      = (RPX * HIDP * 2) > (256 * 64 * 4) ? (RPX * HIDP * 2) : (256 * 64 * 4);
constexpr int SMEM_BYTES = OFF_H + SZ_H;                   // ~290,928 B
static_assert(SMEM_BYTES <= 320 * 1024, "exceeds 320KB WGP LDS");

union FragU { v16bf v; uint32_t u[8]; };

// A-matrix 16x32 bf16 fragment (doc layout: lane<16 holds K{0..7,16..23}, lane>=16 K{8..15,24..31}).
// LDS layout is row-major with K contiguous -> each K-pair is one dword load.
__device__ inline v16bf load_matA(const __bf16* base, int row0, int strideHalves, int k0, int lane) {
  const int r = row0 + (lane & 15);
  const int koff2 = (lane < 16) ? 0 : 4;   // dword offset
  const uint32_t* p = (const uint32_t*)(base + (size_t)r * strideHalves + k0);
  FragU f;
#pragma unroll
  for (int j = 0; j < 4; ++j) f.u[j] = p[j + koff2];
#pragma unroll
  for (int j = 0; j < 4; ++j) f.u[4 + j] = p[8 + j + koff2];
  return f.v;
}

// B-matrix 32x16 bf16 fragment (lane<16 holds K{0..7,16..23}, lane>=16 K{8..15,24..31}; col = lane&15).
__device__ inline v16bf load_matB(const __bf16* base, int n0, int strideHalves, int k0, int lane) {
  const int n = n0 + (lane & 15);
  const int koff2 = (lane < 16) ? 0 : 4;   // dword offset
  const uint32_t* p = (const uint32_t*)(base + (size_t)n * strideHalves + k0);
  FragU f;
#pragma unroll
  for (int j = 0; j < 4; ++j) f.u[j] = p[j + koff2];
#pragma unroll
  for (int j = 0; j < 4; ++j) f.u[4 + j] = p[8 + j + koff2];
  return f.v;
}

__global__ void __launch_bounds__(256) dffn_fused(const float* __restrict__ x,
                                                  const float* __restrict__ Win,
                                                  const float* __restrict__ fftf,
                                                  const float* __restrict__ Wdw,
                                                  const float* __restrict__ Wout,
                                                  float* __restrict__ out) {
  extern __shared__ char smem[];
  __bf16* wbin  = (__bf16*)(smem + OFF_WBIN);
  __bf16* wbout = (__bf16*)(smem + OFF_WBOUT);
  float*  wdw   = (float*) (smem + OFF_WDW);
  float*  ff    = (float*) (smem + OFF_FFT);
  __bf16* xt    = (__bf16*)(smem + OFF_XT);   // region pixels x channels
  __bf16* gated = (__bf16*)(smem + OFF_XT);   // reuse after GEMM1 consumers done
  __bf16* hb    = (__bf16*)(smem + OFF_H);
  float*  outb  = (float*) (smem + OFF_H);    // reuse after conv stage

  const int tid  = threadIdx.x;
  const int lane = tid & 31;
  const int wave = tid >> 5;
  const int b  = blockIdx.z;
  const int y0 = blockIdx.y * TILE;
  const int x0 = blockIdx.x * TILE;

  // ---- Stage 0: weights -> LDS (bf16 for GEMMs, fp32 for conv/fft) ----
  for (int i = tid; i < HIDP * CIN; i += 256) {           // wbin[o][k]
    int o = i / CIN, k = i % CIN;
    wbin[i] = (o < HID) ? (__bf16)Win[o * CIN + k] : (__bf16)0.0f;
  }
  for (int i = tid; i < CIN * HIDP; i += 256) {           // wbout[d][c]
    int d = i / HIDP, c = i % HIDP;
    wbout[i] = (c < HID) ? (__bf16)Wout[d * HID + c] : (__bf16)0.0f;
  }
  for (int i = tid; i < 2 * HID * 9; i += 256) wdw[i] = Wdw[i];
  for (int i = tid; i < HID * 4;     i += 256) ff[i]  = fftf[i];

  // ---- Stage 1: x halo region -> LDS (zero-fill OOB => correct SAME padding & patch zeros) ----
  for (int i = tid; i < CIN * RPX; i += 256) {
    int c = i / RPX, p = i % RPX;
    int gy = y0 - 2 + p / REG;
    int gx = x0 - 2 + p % REG;
    float v = 0.0f;
    if (gy >= 0 && gy < HH && gx >= 0 && gx < WW)
      v = x[(((size_t)b * CIN + c) * HH + gy) * WW + gx];
    xt[(size_t)p * CIN + c] = (__bf16)v;
  }
  __syncthreads();

  // ---- Stage 2: GEMM1 h[400][176] = xt[400][64] * Win^T, bf16 WMMA, K=64 chained ----
  for (int pt = wave; pt < NPT; pt += 8) {                 // wave-uniform, EXEC all ones
    v16bf a0 = load_matA(xt, pt * 16, CIN, 0,  lane);
    v16bf a1 = load_matA(xt, pt * 16, CIN, 32, lane);
    for (int ot = 0; ot < HIDP / 16; ++ot) {
      v16bf b0 = load_matB(wbin, ot * 16, CIN, 0,  lane);
      v16bf b1 = load_matB(wbin, ot * 16, CIN, 32, lane);
      v8f acc = {};
      acc = __builtin_amdgcn_wmma_f32_16x16x32_bf16(false, a0, false, b0, (short)0, acc, false, false);
      acc = __builtin_amdgcn_wmma_f32_16x16x32_bf16(false, a1, false, b1, (short)0, acc, false, false);
      const int n  = ot * 16 + (lane & 15);
      const int rb = pt * 16 + ((lane < 16) ? 0 : 8);
#pragma unroll
      for (int r = 0; r < 8; ++r) hb[(size_t)(rb + r) * HIDP + n] = (__bf16)acc[r];
    }
  }
  __syncthreads();

  // ---- Stage 3: 2x2 patch spectral filter (rfft2 * f -> irfft2 == scaled Hadamard) ----
  for (int it = tid; it < HID * (REG / 2) * (REG / 2); it += 256) {
    int o  = it / 100;
    int pp = it % 100;
    int pi = (pp / 10) * 2, pj = (pp % 10) * 2;
    size_t i00 = (size_t)(pi * REG + pj) * HIDP + o;
    float a = (float)hb[i00], bb = (float)hb[i00 + HIDP * 0 + 1];
    float c = (float)hb[i00 + (size_t)REG * HIDP], d = (float)hb[i00 + (size_t)REG * HIDP + 1];
    float Y00 = (a + bb + c + d) * ff[o * 4 + 0];
    float Y01 = (a - bb + c - d) * ff[o * 4 + 1];
    float Y10 = (a + bb - c - d) * ff[o * 4 + 2];
    float Y11 = (a - bb - c + d) * ff[o * 4 + 3];
    hb[i00]                           = (__bf16)(0.25f * (Y00 + Y01 + Y10 + Y11));
    hb[i00 + 1]                       = (__bf16)(0.25f * (Y00 - Y01 + Y10 - Y11));
    hb[i00 + (size_t)REG * HIDP]      = (__bf16)(0.25f * (Y00 + Y01 - Y10 - Y11));
    hb[i00 + (size_t)REG * HIDP + 1]  = (__bf16)(0.25f * (Y00 - Y01 - Y10 + Y11));
  }
  __syncthreads();

  // ---- Stage 4: depthwise 3x3 conv (gate: h[c>>1], value: h[85+(c>>1)]) + exact GELU gating ----
  for (int it = tid; it < HIDP * 256; it += 256) {
    int c = it / 256, p = it % 256;
    int py = p >> 4, px = p & 15;
    float g = 0.0f;
    if (c < HID) {
      const int g1 = c >> 1;
      const int g2 = 85 + (c >> 1);
      const float* w1 = wdw + (size_t)c * 9;
      const float* w2 = wdw + (size_t)(HID + c) * 9;
      float s1 = 0.0f, s2 = 0.0f;
#pragma unroll
      for (int ky = 0; ky < 3; ++ky)
#pragma unroll
        for (int kx = 0; kx < 3; ++kx) {
          size_t hi = (size_t)((py + 1 + ky) * REG + (px + 1 + kx)) * HIDP;
          s1 += (float)hb[hi + g1] * w1[ky * 3 + kx];
          s2 += (float)hb[hi + g2] * w2[ky * 3 + kx];
        }
      float gel = 0.5f * s1 * (1.0f + erff(s1 * 0.70710678118654752f));
      g = gel * s2;
    }
    gated[(size_t)p * HIDP + c] = (__bf16)g;   // zero-padded cols 170..175
  }
  __syncthreads();

  // ---- Stage 5: GEMM2 out[256][64] = gated[256][176] * Wout^T, K=176 chained ----
#pragma unroll
  for (int pi = 0; pi < 2; ++pi) {
    const int pt = wave * 2 + pi;              // 16 pixel tiles over 8 waves
    v16bf afr[6];
#pragma unroll
    for (int kt = 0; kt < 6; ++kt) afr[kt] = load_matA(gated, pt * 16, HIDP, kt * 32, lane);
#pragma unroll
    for (int dt = 0; dt < 4; ++dt) {
      v8f acc = {};
#pragma unroll
      for (int kt = 0; kt < 6; ++kt) {
        v16bf bfr = load_matB(wbout, dt * 16, HIDP, kt * 32, lane);
        acc = __builtin_amdgcn_wmma_f32_16x16x32_bf16(false, afr[kt], false, bfr, (short)0, acc, false, false);
      }
      const int n  = dt * 16 + (lane & 15);
      const int rb = pt * 16 + ((lane < 16) ? 0 : 8);
#pragma unroll
      for (int r = 0; r < 8; ++r) outb[(size_t)(rb + r) * 64 + n] = acc[r];
    }
  }
  __syncthreads();

  // ---- Stage 6: staged writeback (16-float contiguous runs per (d,row)) ----
  for (int i = tid; i < 64 * 256; i += 256) {
    int d = i >> 8, p = i & 255;
    int py = p >> 4, px = p & 15;
    out[(((size_t)b * 64 + d) * HH + (y0 + py)) * WW + (x0 + px)] = outb[(size_t)p * 64 + d];
  }
}

extern "C" void kernel_launch(void* const* d_in, const int* in_sizes, int n_in,
                              void* d_out, int out_size, void* d_ws, size_t ws_size,
                              hipStream_t stream) {
  (void)in_sizes; (void)n_in; (void)out_size; (void)d_ws; (void)ws_size;
  const float* x    = (const float*)d_in[0];
  const float* Win  = (const float*)d_in[1];
  const float* fftf = (const float*)d_in[2];
  const float* Wdw  = (const float*)d_in[3];
  const float* Wout = (const float*)d_in[4];
  float* out = (float*)d_out;

  // ~284KB dynamic LDS per workgroup (CDNA5 WGP allows up to 320KB)
  (void)hipFuncSetAttribute((const void*)dffn_fused,
                            hipFuncAttributeMaxDynamicSharedMemorySize, SMEM_BYTES);
  dim3 grid(WW / TILE, HH / TILE, NB);
  dffn_fused<<<grid, dim3(256, 1, 1), SMEM_BYTES, stream>>>(x, Win, fftf, Wdw, Wout, out);
}
